// CustomTransformerEncoderMoELayer_44032004719287
// MI455X (gfx1250) — matrix operand order repaired and checked by reading, not compiled, not run on hardware
//
#include <hip/hip_runtime.h>

#define DEV __device__ __forceinline__

typedef __attribute__((ext_vector_type(16))) __bf16 v16bf;
typedef __attribute__((ext_vector_type(8)))  __bf16 v8bf;
typedef __attribute__((ext_vector_type(8)))  float  v8f;

namespace {

constexpr int D_MODEL  = 1024;
constexpr int NHEAD    = 16;
constexpr int HEAD_DIM = 64;
constexpr int FF       = 2048;
constexpr int NEXP     = 8;
constexpr int BATCH    = 8;
constexpr int SEQ      = 512;
constexpr int NTOK     = BATCH * SEQ;   // 4096
constexpr int NSLOT    = 2 * NTOK;      // 8192 (every token routed to exactly 2 experts)

// float -> bf16 round-to-nearest-even (bit-level)
DEV __bf16 f2bf(float f) {
  unsigned u = __builtin_bit_cast(unsigned, f);
  u = u + 0x7FFFu + ((u >> 16) & 1u);
  unsigned short h = (unsigned short)(u >> 16);
  return __builtin_bit_cast(__bf16, h);
}

DEV unsigned pack_bf2(float a, float b) {
  return (unsigned)__builtin_bit_cast(unsigned short, f2bf(a)) |
         ((unsigned)__builtin_bit_cast(unsigned short, f2bf(b)) << 16);
}

// A fragment (16x32 bf16): lane half selects contiguous K block [16h,16h+16)
DEV v16bf load_a_frag(const __bf16* row, int half) {
  v8bf lo = *(const v8bf*)(row + 16 * half);
  v8bf hi = *(const v8bf*)(row + 16 * half + 8);
  v16bf a;
#pragma unroll
  for (int i = 0; i < 8; ++i) { a[i] = lo[i]; a[8 + i] = hi[i]; }
  return a;
}

DEV v8f wmma_bf16(v16bf a, v16bf b, v8f c) {
  return __builtin_amdgcn_wmma_f32_16x16x32_bf16(false, a, false, b, (short)0, c,
                                                 false, false);
}

// One K=32 step of a 128x128 block tile; 8 waves arranged 4(M) x 2(N).
// B tile is stored N-major (Bt[n][k]) so B fragments are contiguous v16bf loads.
DEV void mma_step(__bf16 (*As)[32], __bf16 (*Bt)[32],
                  int wm, int wn, int half, int l16, v8f acc[2][4]) {
  v16bf af[2], bf[4];
#pragma unroll
  for (int mi = 0; mi < 2; ++mi)
    af[mi] = load_a_frag(&As[wm * 32 + mi * 16 + l16][0], half);
#pragma unroll
  for (int ni = 0; ni < 4; ++ni)
    bf[ni] = *(const v16bf*)&Bt[wn * 64 + ni * 16 + l16][16 * half];
#pragma unroll
  for (int mi = 0; mi < 2; ++mi)
#pragma unroll
    for (int ni = 0; ni < 4; ++ni)
      acc[mi][ni] = wmma_bf16(af[mi], bf[ni], acc[mi][ni]);
}

// ---------------------------------------------------------------------------
// Projection GEMM: C[M,N] = A[M,K] * B[K,N] + bias  (fp32 memory, bf16 WMMA)
// Requires M % 128 == 0, N % 128 == 0, K % 32 == 0 (all static shapes here),
// so tile fills are unpredicated straight-line load clauses.
// ---------------------------------------------------------------------------
__global__ __launch_bounds__(256)
void gemm_bias_kernel(int M, int N, int K,
                      const float* __restrict__ A, int lda,
                      const float* __restrict__ B, int ldb,
                      const float* __restrict__ bias,
                      float* __restrict__ C, int ldc) {
  __shared__ __align__(32) __bf16 As[128][32];
  __shared__ __align__(32) __bf16 Bt[128][32];
  const int tid = threadIdx.x;
  const int wid = tid >> 5, lane = tid & 31;
  const int half = lane >> 4, l16 = lane & 15;
  const int wm = wid & 3, wn = wid >> 2;
  const int m0 = blockIdx.y * 128, n0 = blockIdx.x * 128;

  v8f acc[2][4] = {};

  for (int k0 = 0; k0 < K; k0 += 32) {
#pragma unroll
    for (int i = 0; i < 4; ++i) {                 // A tile 128x32, float4 loads
      int chunk = tid + i * 256;
      int r = chunk >> 3, c4 = (chunk & 7) * 4;
      float4 v = *(const float4*)&A[(size_t)(m0 + r) * lda + k0 + c4];
      uint2 p; p.x = pack_bf2(v.x, v.y); p.y = pack_bf2(v.z, v.w);
      *(uint2*)&As[r][c4] = p;
    }
#pragma unroll
    for (int i = 0; i < 4; ++i) {                 // B tile 32x128 -> transposed LDS
      int chunk = tid + i * 256;
      int r = chunk >> 5, c4 = (chunk & 31) * 4;
      float4 v = *(const float4*)&B[(size_t)(k0 + r) * ldb + n0 + c4];
      Bt[c4 + 0][r] = f2bf(v.x);
      Bt[c4 + 1][r] = f2bf(v.y);
      Bt[c4 + 2][r] = f2bf(v.z);
      Bt[c4 + 3][r] = f2bf(v.w);
    }
    if (k0 + 32 < K) {                            // warm next K-slab (global_prefetch_b8)
      __builtin_prefetch(&B[(size_t)(k0 + 32 + (tid >> 7)) * ldb + n0 + (tid & 127)], 0, 1);
      __builtin_prefetch(&A[(size_t)(m0 + (tid >> 1)) * lda + k0 + 32 + (tid & 1) * 16], 0, 1);
    }
    __syncthreads();
    mma_step(As, Bt, wm, wn, half, l16, acc);
    __syncthreads();
  }

#pragma unroll
  for (int mi = 0; mi < 2; ++mi)
#pragma unroll
    for (int ni = 0; ni < 4; ++ni)
#pragma unroll
      for (int vv = 0; vv < 8; ++vv) {
        int gm = m0 + wm * 32 + mi * 16 + vv + 8 * half;
        int gn = n0 + wn * 64 + ni * 16 + l16;
        C[(size_t)gm * ldc + gn] = acc[mi][ni][vv] + bias[gn];
      }
}

// ---------------------------------------------------------------------------
// Flash attention: one block per (b,h, 64-row q tile); 4 waves, 16 q-rows each.
// ---------------------------------------------------------------------------
__global__ __launch_bounds__(128)
void attention_kernel(const float* __restrict__ q, const float* __restrict__ k,
                      const float* __restrict__ v, float* __restrict__ o) {
  __shared__ __align__(32) __bf16 Qs[64][HEAD_DIM];   // [t][dk]
  __shared__ __align__(32) __bf16 Ks[64][HEAD_DIM];   // [t][dk] (= N-major for Q@K^T)
  __shared__ __align__(32) __bf16 Vt[HEAD_DIM][64];   // [dv][t] (N-major for P@V)
  __shared__ __align__(32) __bf16 Ps[4][16][64];

  const int tid = threadIdx.x;
  const int wid = tid >> 5, lane = tid & 31;
  const int half = lane >> 4, l16 = lane & 15;
  const int bh = blockIdx.y;
  const int b = bh / NHEAD, hh = bh % NHEAD;
  const int t0 = blockIdx.x * 64;
  const size_t headoff = (size_t)hh * HEAD_DIM;

#pragma unroll
  for (int i = 0; i < 8; ++i) {                   // Q tile (64x64), float4 loads
    int chunk = tid + i * 128;
    int r = chunk >> 4, c4 = (chunk & 15) * 4;
    float4 qv = *(const float4*)&q[(size_t)(b * SEQ + t0 + r) * D_MODEL + headoff + c4];
    uint2 p; p.x = pack_bf2(qv.x, qv.y); p.y = pack_bf2(qv.z, qv.w);
    *(uint2*)&Qs[r][c4] = p;
  }
  __syncthreads();

  float m_run[8], l_run[8];
#pragma unroll
  for (int vv = 0; vv < 8; ++vv) { m_run[vv] = -3.0e38f; l_run[vv] = 0.f; }
  v8f oacc[4] = {};

  for (int kt = 0; kt < SEQ / 64; ++kt) {
    __syncthreads();
#pragma unroll
    for (int i = 0; i < 8; ++i) {                 // K row-major, V transposed
      int chunk = tid + i * 128;
      int r = chunk >> 4, c4 = (chunk & 15) * 4;
      size_t g = (size_t)(b * SEQ + kt * 64 + r) * D_MODEL + headoff + c4;
      float4 kv = *(const float4*)&k[g];
      uint2 p; p.x = pack_bf2(kv.x, kv.y); p.y = pack_bf2(kv.z, kv.w);
      *(uint2*)&Ks[r][c4] = p;
      float4 vv4 = *(const float4*)&v[g];
      Vt[c4 + 0][r] = f2bf(vv4.x);
      Vt[c4 + 1][r] = f2bf(vv4.y);
      Vt[c4 + 2][r] = f2bf(vv4.z);
      Vt[c4 + 3][r] = f2bf(vv4.w);
    }
    __syncthreads();

    // S = Q @ K^T : B fragment (dk,kj) = Ks[kj][dk] -> contiguous v16bf
    v8f s[4] = {};
#pragma unroll
    for (int kc = 0; kc < 2; ++kc) {
      v16bf af = load_a_frag(&Qs[wid * 16 + l16][kc * 32], half);
#pragma unroll
      for (int ni = 0; ni < 4; ++ni) {
        v16bf bf = *(const v16bf*)&Ks[ni * 16 + l16][kc * 32 + 16 * half];
        s[ni] = wmma_bf16(af, bf, s[ni]);
      }
    }

    // online softmax per row (row = vv + 8*half); xor masks < 16 stay in half
#pragma unroll
    for (int vv = 0; vv < 8; ++vv) {
      float mx = -3.0e38f;
#pragma unroll
      for (int ni = 0; ni < 4; ++ni) { s[ni][vv] *= 0.125f; mx = fmaxf(mx, s[ni][vv]); }
#pragma unroll
      for (int om = 8; om >= 1; om >>= 1) mx = fmaxf(mx, __shfl_xor(mx, om));
      float newm = fmaxf(m_run[vv], mx);
      float corr = __expf(m_run[vv] - newm);
      m_run[vv] = newm;
      float rs = 0.f;
#pragma unroll
      for (int ni = 0; ni < 4; ++ni) {
        float p = __expf(s[ni][vv] - newm);
        s[ni][vv] = p;
        rs += p;
      }
#pragma unroll
      for (int om = 8; om >= 1; om >>= 1) rs += __shfl_xor(rs, om);
      l_run[vv] = l_run[vv] * corr + rs;
#pragma unroll
      for (int ni = 0; ni < 4; ++ni) oacc[ni][vv] *= corr;
      int r = vv + 8 * half;
#pragma unroll
      for (int ni = 0; ni < 4; ++ni) Ps[wid][r][ni * 16 + l16] = f2bf(s[ni][vv]);
    }
    __syncthreads();  // C-layout -> A-layout reshuffle of P through LDS

    // O += P @ V : B fragment from Vt (N-major) -> contiguous v16bf
#pragma unroll
    for (int kc = 0; kc < 2; ++kc) {
      v16bf af = load_a_frag(&Ps[wid][l16][kc * 32], half);
#pragma unroll
      for (int ni = 0; ni < 4; ++ni) {
        v16bf bf = *(const v16bf*)&Vt[ni * 16 + l16][kc * 32 + 16 * half];
        oacc[ni] = wmma_bf16(af, bf, oacc[ni]);
      }
    }
  }

#pragma unroll
  for (int vv = 0; vv < 8; ++vv) {
    float inv = 1.0f / l_run[vv];
    int t = t0 + wid * 16 + vv + 8 * half;
#pragma unroll
    for (int ni = 0; ni < 4; ++ni)
      o[(size_t)(b * SEQ + t) * D_MODEL + headoff + ni * 16 + l16] =
          oacc[ni][vv] * inv;
  }
}

// ---------------------------------------------------------------------------
// residual + LayerNorm (one block of 256 threads per token)
// ---------------------------------------------------------------------------
DEV float block_sum(float s, float* red) {
#pragma unroll
  for (int om = 16; om >= 1; om >>= 1) s += __shfl_xor(s, om);
  int wid = threadIdx.x >> 5, lane = threadIdx.x & 31;
  if (lane == 0) red[wid] = s;
  __syncthreads();
  float t = (lane < 8) ? red[lane] : 0.f;
#pragma unroll
  for (int om = 4; om >= 1; om >>= 1) t += __shfl_xor(t, om);
  t = __shfl(t, 0);
  __syncthreads();
  return t;
}

__global__ __launch_bounds__(256)
void add_ln_kernel(const float* __restrict__ a, const float* __restrict__ r,
                   const float* __restrict__ g, const float* __restrict__ bb,
                   float* __restrict__ out) {
  __shared__ float red[8];
  size_t n = blockIdx.x;
  int tid = threadIdx.x;
  float x[4], s = 0.f;
#pragma unroll
  for (int i = 0; i < 4; ++i) {
    int c = tid + i * 256;
    x[i] = a[n * D_MODEL + c] + r[n * D_MODEL + c];
    s += x[i];
  }
  float mean = block_sum(s, red) * (1.0f / D_MODEL);
  float vs = 0.f;
#pragma unroll
  for (int i = 0; i < 4; ++i) { float d = x[i] - mean; vs += d * d; }
  float var = block_sum(vs, red) * (1.0f / D_MODEL);
  float inv = rsqrtf(var + 1e-5f);
#pragma unroll
  for (int i = 0; i < 4; ++i) {
    int c = tid + i * 256;
    out[n * D_MODEL + c] = (x[i] - mean) * inv * g[c] + bb[c];
  }
}

// ---------------------------------------------------------------------------
// Gate: softmax over 8 experts + top-2 (one wave per token)
// ---------------------------------------------------------------------------
__global__ __launch_bounds__(256)
void gate_kernel(const float* __restrict__ x, const float* __restrict__ gW,
                 const float* __restrict__ gb, int* __restrict__ top_e,
                 float* __restrict__ top_w, int* __restrict__ counts) {
  int wid = threadIdx.x >> 5, lane = threadIdx.x & 31;
  int n = blockIdx.x * 8 + wid;
  float acc[NEXP];
#pragma unroll
  for (int e = 0; e < NEXP; ++e) acc[e] = 0.f;
  for (int i = lane; i < D_MODEL; i += 32) {
    float xv = x[(size_t)n * D_MODEL + i];
#pragma unroll
    for (int e = 0; e < NEXP; ++e) acc[e] += xv * gW[i * NEXP + e];
  }
#pragma unroll
  for (int e = 0; e < NEXP; ++e)
#pragma unroll
    for (int om = 16; om >= 1; om >>= 1) acc[e] += __shfl_xor(acc[e], om);
  if (lane == 0) {
    float mx = -3e38f;
#pragma unroll
    for (int e = 0; e < NEXP; ++e) { acc[e] += gb[e]; mx = fmaxf(mx, acc[e]); }
    float s = 0.f;
#pragma unroll
    for (int e = 0; e < NEXP; ++e) { acc[e] = __expf(acc[e] - mx); s += acc[e]; }
    float invs = 1.f / s;
    int e0 = 0; float w0 = -1.f;
#pragma unroll
    for (int e = 0; e < NEXP; ++e) {
      float gg = acc[e] * invs; acc[e] = gg;
      if (gg > w0) { w0 = gg; e0 = e; }
    }
    int e1 = 0; float w1 = -1.f;
#pragma unroll
    for (int e = 0; e < NEXP; ++e)
      if (e != e0 && acc[e] > w1) { w1 = acc[e]; e1 = e; }
    top_e[n * 2] = e0; top_e[n * 2 + 1] = e1;
    top_w[n * 2] = w0; top_w[n * 2 + 1] = w1;
    atomicAdd(&counts[e0], 1);
    atomicAdd(&counts[e1], 1);
  }
}

__global__ void offsets_kernel(const int* __restrict__ counts,
                               int* __restrict__ offsets, int* __restrict__ cursor) {
  if (threadIdx.x == 0) {
    int s = 0;
    for (int e = 0; e < NEXP; ++e) { offsets[e] = s; cursor[e] = s; s += counts[e]; }
  }
}

__global__ __launch_bounds__(256)
void assign_kernel(const int* __restrict__ top_e, const float* __restrict__ top_w,
                   int* __restrict__ cursor, int* __restrict__ slot_token,
                   float* __restrict__ slot_w) {
  int n = blockIdx.x * 256 + threadIdx.x;
  if (n >= NTOK) return;
#pragma unroll
  for (int j = 0; j < 2; ++j) {
    int e = top_e[n * 2 + j];
    int pos = atomicAdd(&cursor[e], 1);
    slot_token[pos] = n;
    slot_w[pos] = top_w[n * 2 + j];
  }
}

// ---------------------------------------------------------------------------
// MoE GEMM1: h[slot, :FF] = relu(x[token(slot)] @ W1_e + b1_e), h stored bf16
// ---------------------------------------------------------------------------
__global__ __launch_bounds__(256)
void moe_gemm1_kernel(const float* __restrict__ X, const float* __restrict__ W1,
                      const float* __restrict__ b1, __bf16* __restrict__ H,
                      const int* __restrict__ counts, const int* __restrict__ offsets,
                      const int* __restrict__ slot_token) {
  const int e = blockIdx.z;
  const int cnt = counts[e];
  const int m0 = blockIdx.y * 128;
  if (m0 >= cnt) return;
  const int base = offsets[e];
  const int n0 = blockIdx.x * 128;
  const float* B = W1 + (size_t)e * D_MODEL * FF;

  __shared__ __align__(32) __bf16 As[128][32];
  __shared__ __align__(32) __bf16 Bt[128][32];
  const int tid = threadIdx.x;
  const int wid = tid >> 5, lane = tid & 31;
  const int half = lane >> 4, l16 = lane & 15;
  const int wm = wid & 3, wn = wid >> 2;

  v8f acc[2][4] = {};

  for (int k0 = 0; k0 < D_MODEL; k0 += 32) {
#pragma unroll
    for (int i = 0; i < 4; ++i) {                 // gathered A tile, float4 loads
      int chunk = tid + i * 256;
      int r = chunk >> 3, c4 = (chunk & 7) * 4;
      int gr = m0 + r;
      float4 v = {0.f, 0.f, 0.f, 0.f};
      if (gr < cnt) {
        int tok = slot_token[base + gr];
        v = *(const float4*)&X[(size_t)tok * D_MODEL + k0 + c4];
      }
      uint2 p; p.x = pack_bf2(v.x, v.y); p.y = pack_bf2(v.z, v.w);
      *(uint2*)&As[r][c4] = p;
    }
#pragma unroll
    for (int i = 0; i < 4; ++i) {                 // W1 tile -> transposed LDS
      int chunk = tid + i * 256;
      int r = chunk >> 5, c4 = (chunk & 31) * 4;
      float4 v = *(const float4*)&B[(size_t)(k0 + r) * FF + n0 + c4];
      Bt[c4 + 0][r] = f2bf(v.x);
      Bt[c4 + 1][r] = f2bf(v.y);
      Bt[c4 + 2][r] = f2bf(v.z);
      Bt[c4 + 3][r] = f2bf(v.w);
    }
    if (k0 + 32 < D_MODEL)
      __builtin_prefetch(&B[(size_t)(k0 + 32 + (tid >> 7)) * FF + n0 + (tid & 127)], 0, 1);
    __syncthreads();
    mma_step(As, Bt, wm, wn, half, l16, acc);
    __syncthreads();
  }

#pragma unroll
  for (int mi = 0; mi < 2; ++mi)
#pragma unroll
    for (int ni = 0; ni < 4; ++ni)
#pragma unroll
      for (int vv = 0; vv < 8; ++vv) {
        int gmr = m0 + wm * 32 + mi * 16 + vv + 8 * half;
        int gn = n0 + wn * 64 + ni * 16 + l16;
        if (gmr < cnt) {
          float x = acc[mi][ni][vv] + b1[(size_t)e * FF + gn];
          H[(size_t)(base + gmr) * FF + gn] = f2bf(fmaxf(x, 0.f));
        }
      }
}

// ---------------------------------------------------------------------------
// MoE GEMM2: moe_out[token] += w_slot * (h[slot] @ W2_e + b2_e)  (atomic scatter)
// ---------------------------------------------------------------------------
__global__ __launch_bounds__(256)
void moe_gemm2_kernel(const __bf16* __restrict__ H, const float* __restrict__ W2,
                      const float* __restrict__ b2, float* __restrict__ Y,
                      const int* __restrict__ counts, const int* __restrict__ offsets,
                      const int* __restrict__ slot_token,
                      const float* __restrict__ slot_w) {
  const int e = blockIdx.z;
  const int cnt = counts[e];
  const int m0 = blockIdx.y * 128;
  if (m0 >= cnt) return;
  const int base = offsets[e];
  const int n0 = blockIdx.x * 128;
  const float* B = W2 + (size_t)e * FF * D_MODEL;

  __shared__ __align__(32) __bf16 As[128][32];
  __shared__ __align__(32) __bf16 Bt[128][32];
  const int tid = threadIdx.x;
  const int wid = tid >> 5, lane = tid & 31;
  const int half = lane >> 4, l16 = lane & 15;
  const int wm = wid & 3, wn = wid >> 2;

  v8f acc[2][4] = {};

  for (int k0 = 0; k0 < FF; k0 += 32) {
#pragma unroll
    for (int i = 0; i < 2; ++i) {                 // A tile: h already bf16, v8bf loads
      int chunk = tid + i * 256;
      int r = chunk >> 2, c8 = (chunk & 3) * 8;
      int gr = m0 + r;
      v8bf hv = {};
      if (gr < cnt) hv = *(const v8bf*)&H[(size_t)(base + gr) * FF + k0 + c8];
      *(v8bf*)&As[r][c8] = hv;
    }
#pragma unroll
    for (int i = 0; i < 4; ++i) {                 // W2 tile -> transposed LDS
      int chunk = tid + i * 256;
      int r = chunk >> 5, c4 = (chunk & 31) * 4;
      float4 v = *(const float4*)&B[(size_t)(k0 + r) * D_MODEL + n0 + c4];
      Bt[c4 + 0][r] = f2bf(v.x);
      Bt[c4 + 1][r] = f2bf(v.y);
      Bt[c4 + 2][r] = f2bf(v.z);
      Bt[c4 + 3][r] = f2bf(v.w);
    }
    if (k0 + 32 < FF)
      __builtin_prefetch(&B[(size_t)(k0 + 32 + (tid >> 7)) * D_MODEL + n0 + (tid & 127)], 0, 1);
    __syncthreads();
    mma_step(As, Bt, wm, wn, half, l16, acc);
    __syncthreads();
  }

#pragma unroll
  for (int mi = 0; mi < 2; ++mi)
#pragma unroll
    for (int ni = 0; ni < 4; ++ni)
#pragma unroll
      for (int vv = 0; vv < 8; ++vv) {
        int gmr = m0 + wm * 32 + mi * 16 + vv + 8 * half;
        int gn = n0 + wn * 64 + ni * 16 + l16;
        if (gmr < cnt) {
          int tok = slot_token[base + gmr];
          float w = slot_w[base + gmr];
          float val = acc[mi][ni][vv] + b2[(size_t)e * D_MODEL + gn];
          atomicAdd(&Y[(size_t)tok * D_MODEL + gn], w * val);
        }
      }
}

}  // namespace

extern "C" void kernel_launch(void* const* d_in, const int* in_sizes, int n_in,
                              void* d_out, int out_size, void* d_ws, size_t ws_size,
                              hipStream_t stream) {
  (void)in_sizes; (void)n_in; (void)out_size; (void)ws_size;
  const float* src  = (const float*)d_in[0];
  const float* Wq   = (const float*)d_in[1];
  const float* bq   = (const float*)d_in[2];
  const float* Wk   = (const float*)d_in[3];
  const float* bk   = (const float*)d_in[4];
  const float* Wv   = (const float*)d_in[5];
  const float* bv   = (const float*)d_in[6];
  const float* Wo   = (const float*)d_in[7];
  const float* bo   = (const float*)d_in[8];
  const float* gW   = (const float*)d_in[9];
  const float* gb   = (const float*)d_in[10];
  const float* W1   = (const float*)d_in[11];
  const float* b1   = (const float*)d_in[12];
  const float* W2   = (const float*)d_in[13];
  const float* b2   = (const float*)d_in[14];
  const float* ln1g = (const float*)d_in[15];
  const float* ln1b = (const float*)d_in[16];
  const float* ln2g = (const float*)d_in[17];
  const float* ln2b = (const float*)d_in[18];
  float* out = (float*)d_out;

  char* ws = (char*)d_ws;
  size_t off = 0;
  auto alloc = [&](size_t bytes) -> void* {
    off = (off + 255) & ~(size_t)255;
    void* p = ws + off;
    off += bytes;
    return p;
  };
  const size_t actBytes = (size_t)NTOK * D_MODEL * sizeof(float);
  float* q        = (float*)alloc(actBytes);
  float* k        = (float*)alloc(actBytes);
  float* v        = (float*)alloc(actBytes);
  float* attn     = (float*)alloc(actBytes);
  float* attnp    = (float*)alloc(actBytes);
  float* x1       = (float*)alloc(actBytes);
  float* moe_out  = (float*)alloc(actBytes);
  __bf16* hbuf    = (__bf16*)alloc((size_t)NSLOT * FF * sizeof(__bf16));
  int*   top_e    = (int*)alloc((size_t)NTOK * 2 * sizeof(int));
  float* top_w    = (float*)alloc((size_t)NTOK * 2 * sizeof(float));
  int*   counts   = (int*)alloc(NEXP * sizeof(int));
  int*   offsets  = (int*)alloc(NEXP * sizeof(int));
  int*   cursor   = (int*)alloc(NEXP * sizeof(int));
  int*   slot_tok = (int*)alloc((size_t)NSLOT * sizeof(int));
  float* slot_w   = (float*)alloc((size_t)NSLOT * sizeof(float));

  dim3 gProj(D_MODEL / 128, NTOK / 128);  // (8, 32)

  // --- attention block ---
  gemm_bias_kernel<<<gProj, 256, 0, stream>>>(NTOK, D_MODEL, D_MODEL, src, D_MODEL,
                                              Wq, D_MODEL, bq, q, D_MODEL);
  gemm_bias_kernel<<<gProj, 256, 0, stream>>>(NTOK, D_MODEL, D_MODEL, src, D_MODEL,
                                              Wk, D_MODEL, bk, k, D_MODEL);
  gemm_bias_kernel<<<gProj, 256, 0, stream>>>(NTOK, D_MODEL, D_MODEL, src, D_MODEL,
                                              Wv, D_MODEL, bv, v, D_MODEL);
  attention_kernel<<<dim3(SEQ / 64, BATCH * NHEAD), 128, 0, stream>>>(q, k, v, attn);
  gemm_bias_kernel<<<gProj, 256, 0, stream>>>(NTOK, D_MODEL, D_MODEL, attn, D_MODEL,
                                              Wo, D_MODEL, bo, attnp, D_MODEL);
  add_ln_kernel<<<NTOK, 256, 0, stream>>>(src, attnp, ln1g, ln1b, x1);

  // --- MoE block (top-2 routed) ---
  hipMemsetAsync(counts, 0, NEXP * sizeof(int), stream);
  hipMemsetAsync(moe_out, 0, actBytes, stream);
  gate_kernel<<<NTOK / 8, 256, 0, stream>>>(x1, gW, gb, top_e, top_w, counts);
  offsets_kernel<<<1, 32, 0, stream>>>(counts, offsets, cursor);
  assign_kernel<<<NTOK / 256, 256, 0, stream>>>(top_e, top_w, cursor, slot_tok, slot_w);
  moe_gemm1_kernel<<<dim3(FF / 128, NTOK / 128, NEXP), 256, 0, stream>>>(
      x1, W1, b1, hbuf, counts, offsets, slot_tok);
  moe_gemm2_kernel<<<dim3(D_MODEL / 128, NTOK / 128, NEXP), 256, 0, stream>>>(
      hbuf, W2, b2, moe_out, counts, offsets, slot_tok, slot_w);
  add_ln_kernel<<<NTOK, 256, 0, stream>>>(x1, moe_out, ln2g, ln2b, out);
}
